// SwinAttention_12738873000260
// MI455X (gfx1250) — compile-verified, matrix operand order checked by reading, not run on hardware
//
#include <hip/hip_runtime.h>
#include <stdint.h>

typedef __attribute__((ext_vector_type(16))) _Float16 v16h;
typedef __attribute__((ext_vector_type(8)))  float    v8f;

union Frag16 {
  v16h h;
  uint32_t u[8];
  uint4 q[2];
  _Float16 e[16];
};

__device__ __forceinline__ v8f wmma_f16(const Frag16& a, const Frag16& b, v8f c) {
  return __builtin_amdgcn_wmma_f32_16x16x32_f16(false, a.h, false, b.h,
                                                (short)0, c, false, false);
}

// ---------------------------------------------------------------------------
// Kernel: pre-swizzle the four 256x256 fp32 weights into f16 WMMA B-fragments.
// Layout: wfrag[wsel][kt(8)][nt(16)][lane(32)][d(8)] dwords; each dword packs
// W[k0][n], W[k0+1][n] per the 16-bit B-matrix K-swizzle.
// ---------------------------------------------------------------------------
__global__ void k_swizzle_w(const float* __restrict__ Wq, const float* __restrict__ Wk,
                            const float* __restrict__ Wv, const float* __restrict__ Wo,
                            uint32_t* __restrict__ wfrag) {
  int blk  = blockIdx.x;            // 512 = 4 weights * 8 kt * 16 nt
  int wsel = blk >> 7;
  int kt   = (blk >> 4) & 7;
  int nt   = blk & 15;
  const float* W = (wsel == 0) ? Wq : (wsel == 1) ? Wk : (wsel == 2) ? Wv : Wo;
  int tid  = threadIdx.x;           // 256
  int lane = tid >> 3;
  int d    = tid & 7;
  int n     = nt * 16 + (lane & 15);
  int khalf = lane >> 4;
  int k0 = kt * 32 + khalf * 8 + ((d < 4) ? (2 * d) : (16 + 2 * (d - 4)));
  union { uint32_t u; _Float16 h[2]; } p;
  p.h[0] = (_Float16)W[k0 * 256 + n];
  p.h[1] = (_Float16)W[(k0 + 1) * 256 + n];
  wfrag[(size_t)wsel * 32768 + (size_t)((kt * 16 + nt) * 32 + lane) * 8 + d] = p.u;
}

// ---------------------------------------------------------------------------
// Kernel: SwinV2 continuous relative position bias, computed analytically.
// bias16[h][i][j] = 16*sigmoid( relu(table(i,j) @ w1 + b1) @ w2 )[h]
// ---------------------------------------------------------------------------
__global__ void k_cpb(const float* __restrict__ w1, const float* __restrict__ b1,
                      const float* __restrict__ w2, float* __restrict__ bias16) {
  int i = blockIdx.x;    // query token 0..63
  int j = threadIdx.x;   // key token 0..63
  int ri = i >> 3, ci = i & 7, rj = j >> 3, cj = j & 7;
  float a0 = (float)(ri - rj) * (8.0f / 7.0f);
  float a1 = (float)(ci - cj) * (8.0f / 7.0f);
  float t0 = (a0 < 0.f ? -1.f : 1.f) * log2f(fabsf(a0) + 1.f) * (1.f / 3.f);
  float t1 = (a1 < 0.f ? -1.f : 1.f) * log2f(fabsf(a1) + 1.f) * (1.f / 3.f);
  float acc[8] = {0.f, 0.f, 0.f, 0.f, 0.f, 0.f, 0.f, 0.f};
  for (int k = 0; k < 512; ++k) {
    float h = fmaxf(t0 * w1[k] + t1 * w1[512 + k] + b1[k], 0.f);
#pragma unroll
    for (int hh = 0; hh < 8; ++hh) acc[hh] += h * w2[k * 8 + hh];
  }
#pragma unroll
  for (int hh = 0; hh < 8; ++hh)
    bias16[hh * 4096 + i * 64 + j] = 16.f / (1.f + expf(-acc[hh]));
}

// ---------------------------------------------------------------------------
// Kernel: fused roll + window-partition + QKV projection (64x256 @ 256x768).
// One workgroup (8 waves) per window. Pairwise N-tiles + double-buffered
// B-fragment loads so WMMAs overlap with outstanding global_load_b128s.
// Output layout: q/k/v ws[win][head][tok][32] fp32 (unnormalized).
// ---------------------------------------------------------------------------
__global__ void __launch_bounds__(256) k_qkv(
    const float* __restrict__ x, const uint32_t* __restrict__ wfrag,
    const float* __restrict__ bq, const float* __restrict__ bk,
    const float* __restrict__ bv,
    float* __restrict__ qws, float* __restrict__ kws, float* __restrict__ vws) {
  __shared__ _Float16 xw[64][264];   // padded rows: 528B stride, conflict-free
  int wid = blockIdx.x;
  int b = wid >> 6, w64 = wid & 63;
  int wy = w64 >> 3, wx = w64 & 7;
  int tid = threadIdx.x;
  {
    int t = tid >> 2, part = tid & 3;
    int r = t >> 3, c = t & 7;
    int gh = (wy * 8 + r + 4) & 63;          // roll(-4) folded in
    int gw = (wx * 8 + c + 4) & 63;
    const float4* src =
        (const float4*)(x + (((size_t)(b * 64 + gh)) * 64 + gw) * 256 + part * 64);
#pragma unroll
    for (int qq = 0; qq < 16; ++qq) {
      float4 v4 = src[qq];
      int ch = part * 64 + qq * 4;
      xw[t][ch + 0] = (_Float16)v4.x;
      xw[t][ch + 1] = (_Float16)v4.y;
      xw[t][ch + 2] = (_Float16)v4.z;
      xw[t][ch + 3] = (_Float16)v4.w;
    }
  }
  __syncthreads();

  int wave = tid >> 5, lane = tid & 31;
  int strip = wave & 3;      // 16-row strip
  int colh  = wave >> 2;     // which 384-wide half of the 768 outputs
  int m = lane & 15, khalf = lane >> 4;

  Frag16 a[8];               // whole 16x256 A strip held in registers
#pragma unroll
  for (int kt = 0; kt < 8; ++kt) {
    a[kt].q[0] = *(const uint4*)&xw[strip * 16 + m][kt * 32 + khalf * 8];
    a[kt].q[1] = *(const uint4*)&xw[strip * 16 + m][kt * 32 + khalf * 8 + 16];
  }

  for (int ntp = 0; ntp < 12; ++ntp) {       // 12 pairs of 16-wide N-tiles
    int nt0 = ntp * 2;
    int gn  = colh * 384 + nt0 * 16;         // 32-aligned -> pair shares head
    int sel = gn >> 8;                       // 0=q 1=k 2=v
    int ntl = (gn & 255) >> 4;
    const uint32_t* wp0 =
        wfrag + (size_t)sel * 32768 + (size_t)(ntl * 32 + lane) * 8;
    const uint32_t* wp1 = wp0 + 256;         // next N-tile

    v8f acc0 = {}, acc1 = {};
    Frag16 bf0[2], bf1[2];
    bf0[0].q[0] = *(const uint4*)(wp0);
    bf0[0].q[1] = *(const uint4*)(wp0 + 4);
    bf1[0].q[0] = *(const uint4*)(wp1);
    bf1[0].q[1] = *(const uint4*)(wp1 + 4);
#pragma unroll
    for (int kt = 0; kt < 8; ++kt) {
      int cur = kt & 1, nxt = cur ^ 1;
      if (kt < 7) {                          // prefetch next K-step fragments
        const uint32_t* p0 = wp0 + (size_t)(kt + 1) * 4096;
        const uint32_t* p1 = wp1 + (size_t)(kt + 1) * 4096;
        bf0[nxt].q[0] = *(const uint4*)p0;
        bf0[nxt].q[1] = *(const uint4*)(p0 + 4);
        bf1[nxt].q[0] = *(const uint4*)p1;
        bf1[nxt].q[1] = *(const uint4*)(p1 + 4);
      }
      acc0 = wmma_f16(a[kt], bf0[cur], acc0);
      acc1 = wmma_f16(a[kt], bf1[cur], acc1);
    }

    int cbase = gn & 255;                    // multiple of 32
    const float* bias = (sel == 0) ? bq : (sel == 1) ? bk : bv;
    float* dst        = (sel == 0) ? qws : (sel == 1) ? kws : vws;
    int head = cbase >> 5;
    float bb0 = bias[cbase + m];
    float bb1 = bias[cbase + 16 + m];
    float* dbase = dst + (((size_t)wid * 8 + head) * 64) * 32;
#pragma unroll
    for (int i = 0; i < 8; ++i) {
      int tok = strip * 16 + i + 8 * khalf;
      dbase[(size_t)tok * 32 + m]      = acc0[i] + bb0;
      dbase[(size_t)tok * 32 + 16 + m] = acc1[i] + bb1;
    }
  }
}

// ---------------------------------------------------------------------------
// Kernel: per (window, head) cosine attention. 4 waves / block.
// S = qn@kn^T (K=32 -> one WMMA per tile); scale+bias+mask+softmax in LDS;
// O = P@V (two WMMAs per 16x16 tile). O stored f16 as [win][tok][256].
// ---------------------------------------------------------------------------
__global__ void __launch_bounds__(128) k_attn(
    const float* __restrict__ qws, const float* __restrict__ kws,
    const float* __restrict__ vws, const float* __restrict__ logit_scale,
    const float* __restrict__ bias16, _Float16* __restrict__ ows) {
  __shared__ _Float16 qh[64][40];
  __shared__ _Float16 kh[64][40];
  __shared__ _Float16 vT[32][72];    // v transposed: [dim][tok]
  __shared__ _Float16 Ph[64][72];
  __shared__ float    S[64][68];

  int head = blockIdx.x & 7;
  int wid  = blockIdx.x >> 3;
  int wy = (wid & 63) >> 3, wx = wid & 7;
  int tid = threadIdx.x;
  size_t base = ((size_t)wid * 8 + head) * 64 * 32;

  if (tid < 64) {                    // waves 0-1: q rows + v transpose
    const float* qr = qws + base + (size_t)tid * 32;
    const float* vr = vws + base + (size_t)tid * 32;
    float tmp[32]; float ss = 0.f;
#pragma unroll
    for (int d = 0; d < 32; ++d) { tmp[d] = qr[d]; ss += tmp[d] * tmp[d]; }
    float inv = 1.f / fmaxf(sqrtf(ss), 1e-12f);
#pragma unroll
    for (int d = 0; d < 32; ++d) qh[tid][d] = (_Float16)(tmp[d] * inv);
#pragma unroll
    for (int d = 0; d < 32; ++d) vT[d][tid] = (_Float16)vr[d];
  } else {                           // waves 2-3: k rows
    int t = tid - 64;
    const float* kr = kws + base + (size_t)t * 32;
    float tmp[32]; float ss = 0.f;
#pragma unroll
    for (int d = 0; d < 32; ++d) { tmp[d] = kr[d]; ss += tmp[d] * tmp[d]; }
    float inv = 1.f / fmaxf(sqrtf(ss), 1e-12f);
#pragma unroll
    for (int d = 0; d < 32; ++d) kh[t][d] = (_Float16)(tmp[d] * inv);
  }
  __syncthreads();

  int wave = tid >> 5, lane = tid & 31;
  int m = lane & 15, khalf = lane >> 4;
  int strip = wave;                  // each wave owns a 16-row strip of S

  Frag16 a;
  a.q[0] = *(const uint4*)&qh[strip * 16 + m][khalf * 8];
  a.q[1] = *(const uint4*)&qh[strip * 16 + m][khalf * 8 + 16];
#pragma unroll
  for (int nt = 0; nt < 4; ++nt) {
    Frag16 bf;
    bf.q[0] = *(const uint4*)&kh[nt * 16 + m][khalf * 8];
    bf.q[1] = *(const uint4*)&kh[nt * 16 + m][khalf * 8 + 16];
    v8f acc = {};
    acc = wmma_f16(a, bf, acc);
#pragma unroll
    for (int i = 0; i < 8; ++i)
      S[strip * 16 + i + 8 * khalf][nt * 16 + m] = acc[i];
  }
  __syncthreads();

  float scale = expf(fminf(logit_scale[head], 4.6051702f));  // log(100)
  if (tid < 64) {
    int i = tid;
    int hI = wy * 8 + (i >> 3), wI = wx * 8 + (i & 7);
    int regi = ((hI < 56) ? 0 : (hI < 60) ? 1 : 2) * 3 +
               ((wI < 56) ? 0 : (wI < 60) ? 1 : 2);
    const float* brow = bias16 + head * 4096 + i * 64;
    float mx = -1e30f;
    for (int j = 0; j < 64; ++j) {
      int hJ = wy * 8 + (j >> 3), wJ = wx * 8 + (j & 7);
      int regj = ((hJ < 56) ? 0 : (hJ < 60) ? 1 : 2) * 3 +
                 ((wJ < 56) ? 0 : (wJ < 60) ? 1 : 2);
      float val = S[i][j] * scale + brow[j] + ((regi == regj) ? 0.f : -100.f);
      S[i][j] = val;
      mx = fmaxf(mx, val);
    }
    float sum = 0.f;
    for (int j = 0; j < 64; ++j) { float e = expf(S[i][j] - mx); S[i][j] = e; sum += e; }
    float inv = 1.f / sum;
    for (int j = 0; j < 64; ++j) Ph[i][j] = (_Float16)(S[i][j] * inv);
  }
  __syncthreads();

  v8f o0 = {}, o1 = {};
#pragma unroll
  for (int kt = 0; kt < 2; ++kt) {   // K = 64 tokens -> 2 WMMA steps
    Frag16 ap, b0, b1;
    ap.q[0] = *(const uint4*)&Ph[strip * 16 + m][kt * 32 + khalf * 8];
    ap.q[1] = *(const uint4*)&Ph[strip * 16 + m][kt * 32 + khalf * 8 + 16];
    b0.q[0] = *(const uint4*)&vT[m][kt * 32 + khalf * 8];
    b0.q[1] = *(const uint4*)&vT[m][kt * 32 + khalf * 8 + 16];
    b1.q[0] = *(const uint4*)&vT[16 + m][kt * 32 + khalf * 8];
    b1.q[1] = *(const uint4*)&vT[16 + m][kt * 32 + khalf * 8 + 16];
    o0 = wmma_f16(ap, b0, o0);
    o1 = wmma_f16(ap, b1, o1);
  }
  _Float16* ob = ows + ((size_t)wid * 64) * 256 + head * 32;
#pragma unroll
  for (int i = 0; i < 8; ++i) {
    int tok = strip * 16 + i + 8 * khalf;
    ob[(size_t)tok * 256 + m]      = (_Float16)o0[i];
    ob[(size_t)tok * 256 + 16 + m] = (_Float16)o1[i];
  }
}

// ---------------------------------------------------------------------------
// Kernel: output projection (64x256 @ 256x256) + inverse partition + roll(+4).
// Pairwise N-tiles + double-buffered B-fragment loads.
// ---------------------------------------------------------------------------
__global__ void __launch_bounds__(256) k_oproj(
    const _Float16* __restrict__ ows, const uint32_t* __restrict__ wfrag,
    const float* __restrict__ bo, float* __restrict__ out) {
  int wid = blockIdx.x;
  int b = wid >> 6, w64 = wid & 63;
  int wy = w64 >> 3, wx = w64 & 7;
  int tid = threadIdx.x;
  int wave = tid >> 5, lane = tid & 31;
  int strip = wave & 3, nh = wave >> 2;
  int m = lane & 15, khalf = lane >> 4;

  const _Float16* orow = ows + ((size_t)wid * 64 + strip * 16 + m) * 256;
  Frag16 a[8];
#pragma unroll
  for (int kt = 0; kt < 8; ++kt) {
    a[kt].q[0] = *(const uint4*)(orow + kt * 32 + khalf * 8);
    a[kt].q[1] = *(const uint4*)(orow + kt * 32 + khalf * 8 + 16);
  }
  for (int ntp = 0; ntp < 4; ++ntp) {        // 4 pairs of 16-wide N-tiles
    int gn  = nh * 128 + ntp * 32;
    int ntl = gn >> 4;
    const uint32_t* wp0 =
        wfrag + (size_t)3 * 32768 + (size_t)(ntl * 32 + lane) * 8;
    const uint32_t* wp1 = wp0 + 256;

    v8f acc0 = {}, acc1 = {};
    Frag16 bf0[2], bf1[2];
    bf0[0].q[0] = *(const uint4*)(wp0);
    bf0[0].q[1] = *(const uint4*)(wp0 + 4);
    bf1[0].q[0] = *(const uint4*)(wp1);
    bf1[0].q[1] = *(const uint4*)(wp1 + 4);
#pragma unroll
    for (int kt = 0; kt < 8; ++kt) {
      int cur = kt & 1, nxt = cur ^ 1;
      if (kt < 7) {
        const uint32_t* p0 = wp0 + (size_t)(kt + 1) * 4096;
        const uint32_t* p1 = wp1 + (size_t)(kt + 1) * 4096;
        bf0[nxt].q[0] = *(const uint4*)p0;
        bf0[nxt].q[1] = *(const uint4*)(p0 + 4);
        bf1[nxt].q[0] = *(const uint4*)p1;
        bf1[nxt].q[1] = *(const uint4*)(p1 + 4);
      }
      acc0 = wmma_f16(a[kt], bf0[cur], acc0);
      acc1 = wmma_f16(a[kt], bf1[cur], acc1);
    }

    float bb0 = bo[gn + m];
    float bb1 = bo[gn + 16 + m];
#pragma unroll
    for (int i = 0; i < 8; ++i) {
      int tok = strip * 16 + i + 8 * khalf;
      int r = tok >> 3, c = tok & 7;
      int oh = (wy * 8 + r + 4) & 63;        // roll(+4) folded in
      int ow = (wx * 8 + c + 4) & 63;
      float* op = out + (((size_t)(b * 64 + oh)) * 64 + ow) * 256 + gn + m;
      op[0]  = acc0[i] + bb0;
      op[16] = acc1[i] + bb1;
    }
  }
}

extern "C" void kernel_launch(void* const* d_in, const int* in_sizes, int n_in,
                              void* d_out, int out_size, void* d_ws, size_t ws_size,
                              hipStream_t stream) {
  (void)in_sizes; (void)n_in; (void)out_size; (void)ws_size;
  const float* x  = (const float*)d_in[0];
  const float* Wq = (const float*)d_in[1];
  const float* bq = (const float*)d_in[2];
  const float* Wk = (const float*)d_in[3];
  const float* bk = (const float*)d_in[4];
  const float* Wv = (const float*)d_in[5];
  const float* bv = (const float*)d_in[6];
  const float* Wo = (const float*)d_in[7];
  const float* bo = (const float*)d_in[8];
  const float* ls = (const float*)d_in[9];
  const float* w1 = (const float*)d_in[10];
  const float* b1 = (const float*)d_in[11];
  const float* w2 = (const float*)d_in[12];
  float* out = (float*)d_out;

  uint8_t* wsb = (uint8_t*)d_ws;
  uint32_t* wfrag = (uint32_t*)wsb;                              // 512 KB
  float* bias16   = (float*)(wsb + (512u << 10));                // 128 KB
  float* qws      = (float*)(wsb + (512u << 10) + (128u << 10)); // 64 MB each
  float* kws      = qws + (size_t)1024 * 8 * 64 * 32;
  float* vws      = kws + (size_t)1024 * 8 * 64 * 32;
  _Float16* ows   = (_Float16*)(vws + (size_t)1024 * 8 * 64 * 32); // 32 MB

  k_swizzle_w<<<512, 256, 0, stream>>>(Wq, Wk, Wv, Wo, wfrag);
  k_cpb<<<64, 64, 0, stream>>>(w1, b1, w2, bias16);
  k_qkv<<<1024, 256, 0, stream>>>(x, wfrag, bq, bk, bv, qws, kws, vws);
  k_attn<<<8192, 128, 0, stream>>>(qws, kws, vws, ls, bias16, ows);
  k_oproj<<<1024, 256, 0, stream>>>(ows, wfrag, bo, out);
}
